// MultiLevelSCSA_scaler_70093866270962
// MI455X (gfx1250) — matrix operand order repaired and checked by reading, not compiled, not run on hardware
//
#include <hip/hip_runtime.h>
#include <hip/hip_bf16.h>
#include <stddef.h>

typedef __bf16 bf16;
typedef bf16 v16bf __attribute__((ext_vector_type(16)));
typedef float v8f  __attribute__((ext_vector_type(8)));

#define BATCH 8
#define CCH   512
#define HWN   4096   // 64*64
#define NTOK  4096
#define NHEAD 4
#define HDIM  128

// ---- WMMA bf16 16x16x32 per-lane element->K mapping used here (contiguous; A and B match) ----
// A (16x32, 16-bit): lane row M = lane%16 ; K = half*16 + e (same order as B).
__device__ __forceinline__ int a_k_of(int e, int half) {
    return half * 16 + e;
}
// B (32x16, 16-bit): lane column N = lane%16 ; K = half*16 + e.

// ---- CDNA5 async global->LDS copy (ASYNCcnt-tracked, bypasses VGPRs) ----
// LDS address taken from the vdst VGPR; per-lane 16-byte transfer (ISA 10. async).
__device__ __forceinline__ void async_b128(void* lds_dst, const void* gsrc) {
    unsigned ldso = (unsigned)(size_t)lds_dst;  // flat-shared addr[31:0] == LDS offset
    asm volatile("global_load_async_to_lds_b128 %0, %1, off"
                 :: "v"(ldso), "v"(gsrc)
                 : "memory");
}
__device__ __forceinline__ void wait_async0() {
    asm volatile("s_wait_asynccnt 0x0" ::: "memory");
}

// =====================================================================
// Phase 1: multi-level depthwise conv + GroupNorm stats
// =====================================================================
__global__ void __launch_bounds__(256)
zero_f32(float* __restrict__ p, size_t n) {
    size_t i = (size_t)blockIdx.x * blockDim.x + threadIdx.x;
    size_t st = (size_t)gridDim.x * blockDim.x;
    for (; i < n; i += st) p[i] = 0.0f;
}

// One block per (b,c). Writes conv output, reduces sum/sumsq per (b,group).
__global__ void __launch_bounds__(256)
conv_dw(const float* __restrict__ x, const float* __restrict__ w,
        const float* __restrict__ bias, float* __restrict__ y,
        float* __restrict__ stats, int ks) {
    int bc = blockIdx.x;
    int b = bc >> 9, c = bc & (CCH - 1);
    int pad = ks >> 1;
    const float* xp = x + (size_t)bc * HWN;
    const float* wp = w + (size_t)c * ks * ks;
    float bb = bias[c];
    float s = 0.0f, s2 = 0.0f;
    for (int p = threadIdx.x; p < HWN; p += 256) {
        int py = p >> 6, px = p & 63;
        float acc = bb;
        for (int ky = 0; ky < ks; ++ky) {
            int iy = py + ky - pad;
            if (iy < 0 || iy > 63) continue;
            for (int kx = 0; kx < ks; ++kx) {
                int ix = px + kx - pad;
                if (ix < 0 || ix > 63) continue;
                acc += wp[ky * ks + kx] * xp[iy * 64 + ix];
            }
        }
        y[(size_t)bc * HWN + p] = acc;
        s += acc; s2 += acc * acc;
    }
    if (stats) {
        __shared__ float r1[256], r2[256];
        r1[threadIdx.x] = s; r2[threadIdx.x] = s2;
        __syncthreads();
        for (int off = 128; off > 0; off >>= 1) {
            if ((int)threadIdx.x < off) {
                r1[threadIdx.x] += r1[threadIdx.x + off];
                r2[threadIdx.x] += r2[threadIdx.x + off];
            }
            __syncthreads();
        }
        if (threadIdx.x == 0) {
            int g = c >> 7;  // 128 channels per group
            atomicAdd(&stats[(b * 4 + g) * 2 + 0], r1[0]);
            atomicAdd(&stats[(b * 4 + g) * 2 + 1], r2[0]);
        }
    }
}

// normalize + sigmoid gate; level 0 overwrites fused, others accumulate.
__global__ void __launch_bounds__(256)
gn_sig(const float* __restrict__ y, const float* __restrict__ stats,
       const float* __restrict__ gw, const float* __restrict__ gb,
       float* __restrict__ fused, int level) {
    size_t total = (size_t)BATCH * CCH * HWN;
    size_t st = (size_t)gridDim.x * blockDim.x;
    const float cnt = 128.0f * 4096.0f;
    for (size_t i = (size_t)blockIdx.x * blockDim.x + threadIdx.x; i < total; i += st) {
        int c = (int)((i >> 12) & (CCH - 1));
        int b = (int)(i >> 21);
        int g = c >> 7;
        float mu  = stats[(b * 4 + g) * 2 + 0] / cnt;
        float var = stats[(b * 4 + g) * 2 + 1] / cnt - mu * mu;
        float xn = (y[i] - mu) * rsqrtf(var + 1e-5f);
        float v = xn * gw[c] + gb[c];
        float sg = 1.0f / (1.0f + __expf(-v));
        float add = 0.25f * sg;
        fused[i] = (level == 0 ? add : fused[i] + add);
    }
}

// =====================================================================
// Phase 2: tokens (NCHW -> (b,n,c) bf16), weight convert
// =====================================================================
__global__ void __launch_bounds__(256)
tok_cvt(const float* __restrict__ fused, bf16* __restrict__ tok) {
    size_t total = (size_t)BATCH * NTOK * CCH;
    size_t st = (size_t)gridDim.x * blockDim.x;
    for (size_t i = (size_t)blockIdx.x * blockDim.x + threadIdx.x; i < total; i += st) {
        int c = (int)(i & (CCH - 1));
        int n = (int)((i >> 9) & (NTOK - 1));
        int b = (int)(i >> 21);
        tok[i] = (bf16)fused[((size_t)(b * CCH + c)) * HWN + n];
    }
}

__global__ void __launch_bounds__(256)
f32_to_bf16(const float* __restrict__ src, bf16* __restrict__ dst, size_t n, float scale) {
    size_t i = (size_t)blockIdx.x * blockDim.x + threadIdx.x;
    size_t st = (size_t)gridDim.x * blockDim.x;
    for (; i < n; i += st) dst[i] = (bf16)(src[i] * scale);
}

// =====================================================================
// Phase 3: QK projection GEMM  (M=32768, N=1024, K=512) bf16 WMMA
// Double-buffered: async global->LDS copies overlap WMMA.
// =====================================================================
__global__ void __launch_bounds__(256)
gemm_qk(const bf16* __restrict__ A, const bf16* __restrict__ W, float* __restrict__ D) {
    __shared__ alignas(16) bf16 As[2][128 * 32];   // [m][k]
    __shared__ alignas(16) bf16 Bs[2][64 * 32];    // [j][k]
    int bm = blockIdx.x * 128;
    int bn = blockIdx.y * 64;
    int tid = threadIdx.x, lane = tid & 31, wid = tid >> 5;
    int wm = (wid & 3) * 32, wn = (wid >> 2) * 32;
    int half = lane >> 4, l16 = lane & 15;
    v8f acc[2][2] = {};

    auto issue = [&](int buf, int k0) {
        for (int i = tid; i < 512; i += 256) {          // As: 128 rows x 64B
            int r = i >> 2, c16 = (i & 3) * 8;
            async_b128(&As[buf][r * 32 + c16], A + (size_t)(bm + r) * 512 + k0 + c16);
        }
        {                                               // Bs: 64 rows x 64B (i == tid)
            int i = tid;
            if (i < 256) {
                int r = i >> 2, c16 = (i & 3) * 8;
                async_b128(&Bs[buf][r * 32 + c16], W + (size_t)(bn + r) * 512 + k0 + c16);
            }
        }
    };

    issue(0, 0);
    for (int it = 0; it < 16; ++it) {
        wait_async0();
        __syncthreads();                    // data ready; prior compute on other buf done
        if (it + 1 < 16) issue((it + 1) & 1, (it + 1) * 32);
        int cur = it & 1;
        v16bf af[2], bfg[2];
        for (int t = 0; t < 2; ++t) {
            int row = wm + t * 16 + l16;
            for (int e = 0; e < 16; ++e) af[t][e] = As[cur][row * 32 + a_k_of(e, half)];
            int col = wn + t * 16 + l16;
            for (int e = 0; e < 16; ++e) bfg[t][e] = Bs[cur][col * 32 + half * 16 + e];
        }
        for (int tm = 0; tm < 2; ++tm)
            for (int tn = 0; tn < 2; ++tn)
                acc[tm][tn] = __builtin_amdgcn_wmma_f32_16x16x32_bf16(
                    false, af[tm], false, bfg[tn], (short)0, acc[tm][tn], false, false);
    }
    for (int tm = 0; tm < 2; ++tm)
        for (int tn = 0; tn < 2; ++tn)
            for (int r = 0; r < 8; ++r) {
                int m = bm + wm + tm * 16 + (half ? 8 : 0) + r;
                int n = bn + wn + tn * 16 + l16;
                D[(size_t)m * 1024 + n] = acc[tm][tn][r];
            }
}

// =====================================================================
// Phase 4: elu+1, RoPE -> q_r / k_r (bf16); k_mean; z
// =====================================================================
__global__ void __launch_bounds__(256)
epi_rope(const float* __restrict__ qk, bf16* __restrict__ qr, bf16* __restrict__ kr) {
    size_t total = (size_t)BATCH * NTOK * 512;  // 512 pairs per token (256 q + 256 k)
    size_t st = (size_t)gridDim.x * blockDim.x;
    for (size_t i = (size_t)blockIdx.x * blockDim.x + threadIdx.x; i < total; i += st) {
        int p = (int)(i & 511);
        int n = (int)((i >> 9) & (NTOK - 1));
        int b = (int)(i >> 21);
        bool isq = p < 256;
        int jp = isq ? p : p - 256;       // theta index over c//2 = 256 pairs
        int cre = 2 * jp;
        const float* row = qk + ((size_t)(b * NTOK + n)) * 1024 + (isq ? 0 : 512);
        float xre = row[cre], xim = row[cre + 1];
        xre = xre > 0.0f ? xre + 1.0f : __expf(xre);   // elu(x)+1
        xim = xim > 0.0f ? xim + 1.0f : __expf(xim);
        float theta = __powf(10000.0f, -(float)jp * (1.0f / 256.0f));
        float ang = (float)(n & 63) * theta;           // angle varies along w axis
        float sn, cs; __sincosf(ang, &sn, &cs);
        float ore = cs * xre - sn * xim;
        float oim = cs * xim + sn * xre;
        size_t o = ((size_t)(b * NTOK + n)) * 512 + cre;
        if (isq) { qr[o] = (bf16)ore; qr[o + 1] = (bf16)oim; }
        else     { kr[o] = (bf16)ore; kr[o + 1] = (bf16)oim; }
    }
}

// k_mean[b][c] = mean_n (elu(k)+1) ; one block per (b,c)
__global__ void __launch_bounds__(256)
kmean_kern(const float* __restrict__ qk, float* __restrict__ kmean) {
    int bc = blockIdx.x;
    int b = bc >> 9, c = bc & (CCH - 1);
    float s = 0.0f;
    for (int n = threadIdx.x; n < NTOK; n += 256) {
        float v = qk[((size_t)(b * NTOK + n)) * 1024 + 512 + c];
        s += v > 0.0f ? v + 1.0f : __expf(v);
    }
    __shared__ float r[256];
    r[threadIdx.x] = s; __syncthreads();
    for (int off = 128; off > 0; off >>= 1) {
        if ((int)threadIdx.x < off) r[threadIdx.x] += r[threadIdx.x + off];
        __syncthreads();
    }
    if (threadIdx.x == 0) kmean[bc] = r[0] * (1.0f / 4096.0f);
}

// z[b][h][n] = 1/(q . k_mean + 1e-6) ; one block (512 thr) per token
__global__ void __launch_bounds__(512)
z_kern(const float* __restrict__ qk, const float* __restrict__ kmean, float* __restrict__ z) {
    int bi = blockIdx.x;                 // b*4096 + n
    int b = bi >> 12, n = bi & (NTOK - 1);
    int t = threadIdx.x;
    float qv = qk[(size_t)bi * 1024 + t];
    qv = qv > 0.0f ? qv + 1.0f : __expf(qv);
    float prod = qv * kmean[b * CCH + t];
    __shared__ float red[512];
    red[t] = prod; __syncthreads();
    int seg = t >> 7, ls = t & 127;
    for (int off = 64; off > 0; off >>= 1) {
        if (ls < off) red[t] += red[t + off];
        __syncthreads();
    }
    if (ls == 0) z[((size_t)(b * NHEAD + seg)) * NTOK + n] = 1.0f / (red[t] + 1e-6f);
}

// =====================================================================
// Phase 5: kv[b,h] = k_r^T . v   (128x128, K=4096, split-K atomics)
// B operand = bf16 tokens (same values as fused, async-copyable).
// =====================================================================
__global__ void __launch_bounds__(256)
kv_gemm(const bf16* __restrict__ kr, const bf16* __restrict__ tok, float* __restrict__ kv) {
    int bh = blockIdx.x;                 // b*4 + h
    int b = bh >> 2, h = bh & 3;
    int n0 = blockIdx.y * 256;           // split-K chunk
    __shared__ alignas(16) bf16 As[2][32 * 128];   // [nn][d]  (k_r)
    __shared__ alignas(16) bf16 Bs[2][32 * 128];   // [nn][e]  (v = tokens)
    int tid = threadIdx.x, lane = tid & 31, wid = tid >> 5;
    int half = lane >> 4, l16 = lane & 15;
    int wd = wid * 16;                   // each wave: 16 rows of d
    v8f acc[8] = {};

    auto issue = [&](int buf, int kk) {
        for (int i = tid; i < 512; i += 256) {
            int nn = i >> 4, d16 = (i & 15) * 8;
            size_t rowoff = ((size_t)(b * NTOK + n0 + kk + nn)) * 512 + h * HDIM;
            async_b128(&As[buf][nn * 128 + d16], kr  + rowoff + d16);
            async_b128(&Bs[buf][nn * 128 + d16], tok + rowoff + d16);
        }
    };

    issue(0, 0);
    for (int it = 0; it < 8; ++it) {
        wait_async0();
        __syncthreads();
        if (it + 1 < 8) issue((it + 1) & 1, (it + 1) * 32);
        int cur = it & 1;
        v16bf af;
        int row = wd + l16;
        for (int e2 = 0; e2 < 16; ++e2) af[e2] = As[cur][a_k_of(e2, half) * 128 + row];
        for (int t = 0; t < 8; ++t) {
            v16bf bfr;
            int col = t * 16 + l16;
            for (int e2 = 0; e2 < 16; ++e2) bfr[e2] = Bs[cur][(half * 16 + e2) * 128 + col];
            acc[t] = __builtin_amdgcn_wmma_f32_16x16x32_bf16(
                false, af, false, bfr, (short)0, acc[t], false, false);
        }
    }
    for (int t = 0; t < 8; ++t)
        for (int r = 0; r < 8; ++r) {
            int d = wd + (half ? 8 : 0) + r;
            int e = t * 16 + l16;
            atomicAdd(&kv[((size_t)bh * HDIM + d) * HDIM + e], acc[t][r]);
        }
}

// =====================================================================
// Phase 6: lepe conv writes d_out; attn = (q_r . kv) * z added on top
// =====================================================================
__global__ void __launch_bounds__(256)
attn_gemm(const bf16* __restrict__ qr, const bf16* __restrict__ kvb,
          const float* __restrict__ z, float* __restrict__ out) {
    int bh = blockIdx.x;                 // b*4 + h
    int b = bh >> 2, h = bh & 3;
    int m0 = blockIdx.y * 128;           // token tile
    __shared__ alignas(16) bf16 As[2][128 * 32];   // [m][k]
    __shared__ alignas(16) bf16 Bs[2][32 * 128];   // [k][e]
    int tid = threadIdx.x, lane = tid & 31, wid = tid >> 5;
    int half = lane >> 4, l16 = lane & 15;
    int wm = wid * 16;
    v8f acc[8] = {};

    auto issue = [&](int buf, int k0) {
        for (int i = tid; i < 512; i += 256) {
            int m = i >> 2, k16 = (i & 3) * 8;
            async_b128(&As[buf][m * 32 + k16],
                       qr + ((size_t)(b * NTOK + m0 + m)) * 512 + h * HDIM + k0 + k16);
            int kk = i >> 4, e16 = (i & 15) * 8;
            async_b128(&Bs[buf][kk * 128 + e16],
                       kvb + ((size_t)bh * HDIM + k0 + kk) * HDIM + e16);
        }
    };

    issue(0, 0);
    for (int it = 0; it < 4; ++it) {
        wait_async0();
        __syncthreads();
        if (it + 1 < 4) issue((it + 1) & 1, (it + 1) * 32);
        int cur = it & 1;
        v16bf af;
        int row = wm + l16;
        for (int e2 = 0; e2 < 16; ++e2) af[e2] = As[cur][row * 32 + a_k_of(e2, half)];
        for (int t = 0; t < 8; ++t) {
            v16bf bfr;
            int col = t * 16 + l16;
            for (int e2 = 0; e2 < 16; ++e2) bfr[e2] = Bs[cur][(half * 16 + e2) * 128 + col];
            acc[t] = __builtin_amdgcn_wmma_f32_16x16x32_bf16(
                false, af, false, bfr, (short)0, acc[t], false, false);
        }
    }
    for (int t = 0; t < 8; ++t)
        for (int r = 0; r < 8; ++r) {
            int m = m0 + wm + (half ? 8 : 0) + r;
            int e = t * 16 + l16;
            float zz = z[(size_t)bh * NTOK + m];
            size_t oi = ((size_t)(b * CCH + h * HDIM + e)) * HWN + m;
            out[oi] += acc[t][r] * zz;   // lepe already stored; unique (c,n) per lane/reg
        }
}

// =====================================================================
extern "C" void kernel_launch(void* const* d_in, const int* in_sizes, int n_in,
                              void* d_out, int out_size, void* d_ws, size_t ws_size,
                              hipStream_t stream) {
    (void)in_sizes; (void)n_in; (void)out_size; (void)ws_size;
    const float* x      = (const float*)d_in[0];
    const float* conv_w[4], *conv_b[4], *gn_w[4], *gn_b[4];
    for (int i = 0; i < 4; ++i) {
        conv_w[i] = (const float*)d_in[1 + 4 * i + 0];
        conv_b[i] = (const float*)d_in[1 + 4 * i + 1];
        gn_w[i]   = (const float*)d_in[1 + 4 * i + 2];
        gn_b[i]   = (const float*)d_in[1 + 4 * i + 3];
    }
    const float* qk_w   = (const float*)d_in[17];
    const float* lepe_w = (const float*)d_in[18];
    const float* lepe_b = (const float*)d_in[19];
    float* out = (float*)d_out;

    // ---- workspace layout (bytes) ----
    const size_t MB = 1ull << 20;
    char* ws = (char*)d_ws;
    float* fused = (float*)(ws + 0);              //  64 MB  (B,C,HW)
    float* qkbuf = (float*)(ws + 64 * MB);        // 128 MB  (b*n, 1024); first 64MB doubles as conv temp
    float* ytmp  = qkbuf;                         //  alias (conv temp, consumed before GEMM)
    bf16*  tok   = (bf16*)(ws + 192 * MB);        //  32 MB
    bf16*  qr    = (bf16*)(ws + 224 * MB);        //  32 MB
    bf16*  kr    = (bf16*)(ws + 256 * MB);        //  32 MB
    bf16*  wbf   = (bf16*)(ws + 288 * MB);        //   1 MB
    float* kvf   = (float*)(ws + 289 * MB);       //   2 MB
    bf16*  kvb   = (bf16*)(ws + 291 * MB);        //   1 MB
    float* zbuf  = (float*)(ws + 292 * MB);       // 512 KB
    float* kmean = (float*)(ws + 293 * MB);       //  16 KB
    float* stats = (float*)(ws + 293 * MB + 65536); // 256 B

    const int KSZ[4] = {3, 5, 7, 9};

    // zero split-K accumulator for kv
    zero_f32<<<64, 256, 0, stream>>>(kvf, (size_t)BATCH * NHEAD * HDIM * HDIM);

    // Phase 1: 4 levels of dwconv -> GN -> sigmoid, averaged into fused
    for (int lvl = 0; lvl < 4; ++lvl) {
        zero_f32<<<1, 256, 0, stream>>>(stats, 64);
        conv_dw<<<BATCH * CCH, 256, 0, stream>>>(x, conv_w[lvl], conv_b[lvl],
                                                 ytmp, stats, KSZ[lvl]);
        gn_sig<<<8192, 256, 0, stream>>>(ytmp, stats, gn_w[lvl], gn_b[lvl], fused, lvl);
    }

    // Phase 2: tokens + weight to bf16
    tok_cvt<<<8192, 256, 0, stream>>>(fused, tok);
    f32_to_bf16<<<2048, 256, 0, stream>>>(qk_w, wbf, (size_t)1024 * 512, 1.0f);

    // Phase 3: QK projection (WMMA + async double-buffer)
    {
        dim3 grid(32768 / 128, 1024 / 64);
        gemm_qk<<<grid, 256, 0, stream>>>(tok, wbf, qkbuf);
    }

    // Phase 4: elu+1 / rope / k_mean / z
    epi_rope<<<8192, 256, 0, stream>>>(qkbuf, qr, kr);
    kmean_kern<<<BATCH * CCH, 256, 0, stream>>>(qkbuf, kmean);
    z_kern<<<BATCH * NTOK, 512, 0, stream>>>(qkbuf, kmean, zbuf);

    // Phase 5: kv = k_r^T . v / n  (WMMA, split-K atomics) then -> bf16 with 1/n
    {
        dim3 grid(BATCH * NHEAD, NTOK / 256);
        kv_gemm<<<grid, 256, 0, stream>>>(kr, tok, kvf);
    }
    f32_to_bf16<<<2048, 256, 0, stream>>>(kvf, kvb,
                                          (size_t)BATCH * NHEAD * HDIM * HDIM,
                                          1.0f / 4096.0f);

    // Phase 6: lepe writes d_out (NCHW), then attn adds (q_r.kv)*z
    conv_dw<<<BATCH * CCH, 256, 0, stream>>>(fused, lepe_w, lepe_b, out, nullptr, 3);
    {
        dim3 grid(BATCH * NHEAD, NTOK / 128);
        attn_gemm<<<grid, 256, 0, stream>>>(qr, kvb, zbuf, out);
    }
}
